// GNNLayer_38852274159909
// MI455X (gfx1250) — matrix operand order repaired and checked by reading, not compiled, not run on hardware
//
#include <hip/hip_runtime.h>
#include <hip/hip_bf16.h>

typedef __bf16 bf16_t;
typedef __attribute__((ext_vector_type(16))) __bf16 v16bf;
typedef __attribute__((ext_vector_type(4)))  __bf16 v4bf;
typedef __attribute__((ext_vector_type(8)))  float  v8f;
typedef __attribute__((ext_vector_type(4)))  float  v4f;
typedef __attribute__((ext_vector_type(4)))  int    v4i;

#define NROW  8192
#define FDIM  128
#define C0DIM 256
#define C1DIM 128
#define BN_EPS 1e-3f

#define PCHUNK 128
#define NCHUNK (NROW / PCHUNK)   // 64 partial chunks for the softmax stats

union FragU { v16bf v; unsigned u[8]; };

// ---------------------------------------------------------------------------
// Masked-softmax row stats, phase 1: online (max, sumexp) partials per
// (4-column group, 128-row p-chunk). 131072 threads -> enough loads in
// flight to saturate HBM; int4 loads = 512B per wave per instruction.
// ---------------------------------------------------------------------------
__global__ void __launch_bounds__(256) softmax_stats_phase1(
    const int* __restrict__ adj, float* __restrict__ pm, float* __restrict__ ps)
{
    int gid   = blockIdx.x * blockDim.x + threadIdx.x;
    int r4    = (gid % (NROW / 4)) * 4;
    int chunk = gid / (NROW / 4);
    int p0    = chunk * PCHUNK;

    float m[4] = {-3.0e38f, -3.0e38f, -3.0e38f, -3.0e38f};
    float s[4] = {0.f, 0.f, 0.f, 0.f};
    for (int i = 0; i < PCHUNK; ++i) {
        v4i v = __builtin_nontemporal_load((const v4i*)&adj[(p0 + i) * NROW + r4]);
#pragma unroll
        for (int j = 0; j < 4; ++j) {
            int vj = v[j];
            if (vj != 0) {
                float f = (float)vj;
                if (f > m[j]) { s[j] = s[j] * __expf(m[j] - f) + 1.0f; m[j] = f; }
                else          { s[j] += __expf(f - m[j]); }
            }
        }
    }
#pragma unroll
    for (int j = 0; j < 4; ++j) {
        pm[chunk * NROW + r4 + j] = m[j];
        ps[chunk * NROW + r4 + j] = s[j];
    }
}

// Phase 2: deterministic fixed-order merge of the 64 chunk partials.
__global__ void __launch_bounds__(256) softmax_stats_phase2(
    const float* __restrict__ pm, const float* __restrict__ ps,
    float* __restrict__ mrow, float* __restrict__ sinv)
{
    int r = blockIdx.x * blockDim.x + threadIdx.x;
    if (r >= NROW) return;
    float M = -3.0e38f, S = 0.0f;
    for (int c = 0; c < NCHUNK; ++c) {
        float m2 = pm[c * NROW + r];
        float s2 = ps[c * NROW + r];
        if (m2 > M) { S = S * __expf(M - m2) + s2; M = m2; }
        else        { S += s2 * __expf(m2 - M); }
    }
    mrow[r] = (S > 0.0f) ? M : 0.0f;
    sinv[r] = (S > 0.0f) ? 1.0f / S : 0.0f;   // empty row -> w row all zeros
}

// ---------------------------------------------------------------------------
// Fused SpMM: C(8192 x NT*16) = W(T) @ B, W built on the fly from adj + stats,
// bf16 in LDS, v_wmma_f32_16x16x32_bf16.
//   WT=false: A[m][k] = f(adj[k*N+m], stats[m])   (C = w   @ B)
//   WT=true : A[m][k] = f(adj[m*N+k], stats[k])   (C = w^T @ B)
// Block: 128 threads = 4 waves; M-tile 64 (16 rows/wave); K-step 32.
// 128 blocks -> fills the WGP array; small LDS lets blocks co-reside.
// ---------------------------------------------------------------------------
template <int NT, bool WT>
__global__ void __launch_bounds__(128) spmm_softmax_wmma(
    const int* __restrict__ adj, const float* __restrict__ mrow,
    const float* __restrict__ sinv, const bf16_t* __restrict__ Bmat,
    float* __restrict__ Cout)
{
    constexpr int NCOL = NT * 16;
    constexpr int LDAB = 36;                      // even stride -> aligned pairs
    __shared__ __align__(16) bf16_t aT[64][LDAB];
    __shared__ __align__(16) bf16_t bT[NCOL][LDAB];

    const int m0   = blockIdx.x * 64;
    const int lane = threadIdx.x & 31;
    const int wv   = threadIdx.x >> 5;
    const int half = lane >> 4;
    const int mr   = lane & 15;

    v8f acc[NT] = {};

    for (int k0 = 0; k0 < NROW; k0 += 32) {
        // ---- stage A tile (64 x 32) of softmax weights, int4 adj loads ----
        for (int idx = threadIdx.x; idx < 64 * 32 / 4; idx += 128) {
            if (WT) {
                int kk = (idx & 7) * 4;
                int mm = idx >> 3;
                v4i v = __builtin_nontemporal_load(
                    (const v4i*)&adj[(m0 + mm) * NROW + (k0 + kk)]);
                v4bf w;
#pragma unroll
                for (int j = 0; j < 4; ++j) {
                    int si = k0 + kk + j;
                    float f = 0.0f;
                    if (v[j] != 0) f = __expf((float)v[j] - mrow[si]) * sinv[si];
                    w[j] = (bf16_t)f;
                }
                *(v4bf*)&aT[mm][kk] = w;          // one ds_store_b64
            } else {
                int mm = (idx & 15) * 4;
                int kk = idx >> 4;
                v4i v = __builtin_nontemporal_load(
                    (const v4i*)&adj[(k0 + kk) * NROW + (m0 + mm)]);
#pragma unroll
                for (int j = 0; j < 4; ++j) {
                    int si = m0 + mm + j;
                    float f = 0.0f;
                    if (v[j] != 0) f = __expf((float)v[j] - mrow[si]) * sinv[si];
                    aT[mm + j][kk] = (bf16_t)f;
                }
            }
        }
        // ---- stage B tile (32 x NCOL) transposed to [n][k], b64 loads ----
        for (int idx = threadIdx.x; idx < 32 * NCOL / 4; idx += 128) {
            int nn = (idx & (NCOL / 4 - 1)) * 4;
            int kk = idx / (NCOL / 4);
            v4bf b = *(const v4bf*)&Bmat[(k0 + kk) * NCOL + nn];
#pragma unroll
            for (int j = 0; j < 4; ++j) bT[nn + j][kk] = b[j];
        }
        __syncthreads();

        FragU a;                                   // A 16x32 bf16 fragment
#pragma unroll
        for (int v = 0; v < 8; ++v) {
            int ka = ((v & 3) * 2) + ((v >> 2) * 16) + half * 8;
            a.u[v] = *(const unsigned*)&aT[wv * 16 + mr][ka];
        }
#pragma unroll
        for (int nt = 0; nt < NT; ++nt) {
            FragU b;                               // B 32x16 bf16 fragment
#pragma unroll
            for (int v = 0; v < 8; ++v) {
                int kb = ((v & 3) * 2) + ((v >> 2) * 16) + half * 8;
                b.u[v] = *(const unsigned*)&bT[nt * 16 + mr][kb];
            }
            acc[nt] = __builtin_amdgcn_wmma_f32_16x16x32_bf16(
                false, a.v, false, b.v, (short)0, acc[nt], false, false);
        }
        __syncthreads();
    }

#pragma unroll
    for (int nt = 0; nt < NT; ++nt)
#pragma unroll
        for (int j = 0; j < 8; ++j) {
            int row = m0 + wv * 16 + half * 8 + j;
            int col = nt * 16 + mr;
            Cout[row * NCOL + col] = acc[nt][j];
        }
}

// ---------------------------------------------------------------------------
// Dense projection: C(8192 x NT*16) = A(f32) @ B(bf16) + bias, same tiling.
// ---------------------------------------------------------------------------
template <int NT>
__global__ void __launch_bounds__(128) gemm_bias_wmma(
    const float* __restrict__ A, const bf16_t* __restrict__ Bmat,
    const float* __restrict__ bias, float* __restrict__ Cout, int Ktot)
{
    constexpr int NCOL = NT * 16;
    constexpr int LDAB = 36;
    __shared__ __align__(16) bf16_t aT[64][LDAB];
    __shared__ __align__(16) bf16_t bT[NCOL][LDAB];

    const int m0   = blockIdx.x * 64;
    const int lane = threadIdx.x & 31;
    const int wv   = threadIdx.x >> 5;
    const int half = lane >> 4;
    const int mr   = lane & 15;

    v8f acc[NT] = {};

    for (int k0 = 0; k0 < Ktot; k0 += 32) {
        for (int idx = threadIdx.x; idx < 64 * 32 / 4; idx += 128) {
            int kk = (idx & 7) * 4;
            int mm = idx >> 3;
            v4f v = *(const v4f*)&A[(m0 + mm) * Ktot + (k0 + kk)];
            v4bf w;
#pragma unroll
            for (int j = 0; j < 4; ++j) w[j] = (bf16_t)v[j];
            *(v4bf*)&aT[mm][kk] = w;
        }
        for (int idx = threadIdx.x; idx < 32 * NCOL / 4; idx += 128) {
            int nn = (idx & (NCOL / 4 - 1)) * 4;
            int kk = idx / (NCOL / 4);
            v4bf b = *(const v4bf*)&Bmat[(k0 + kk) * NCOL + nn];
#pragma unroll
            for (int j = 0; j < 4; ++j) bT[nn + j][kk] = b[j];
        }
        __syncthreads();

        FragU a;
#pragma unroll
        for (int v = 0; v < 8; ++v) {
            int ka = ((v & 3) * 2) + ((v >> 2) * 16) + half * 8;
            a.u[v] = *(const unsigned*)&aT[wv * 16 + mr][ka];
        }
#pragma unroll
        for (int nt = 0; nt < NT; ++nt) {
            FragU b;
#pragma unroll
            for (int v = 0; v < 8; ++v) {
                int kb = ((v & 3) * 2) + ((v >> 2) * 16) + half * 8;
                b.u[v] = *(const unsigned*)&bT[nt * 16 + mr][kb];
            }
            acc[nt] = __builtin_amdgcn_wmma_f32_16x16x32_bf16(
                false, a.v, false, b.v, (short)0, acc[nt], false, false);
        }
        __syncthreads();
    }

#pragma unroll
    for (int nt = 0; nt < NT; ++nt)
#pragma unroll
        for (int j = 0; j < 8; ++j) {
            int row = m0 + wv * 16 + half * 8 + j;
            int col = nt * 16 + mr;
            Cout[row * NCOL + col] = acc[nt][j] + bias[col];
        }
}

// ---------------------------------------------------------------------------
// BN helpers
// ---------------------------------------------------------------------------
__global__ void zero_f32(float* __restrict__ p, int n)
{
    int i = blockIdx.x * blockDim.x + threadIdx.x;
    if (i < n) p[i] = 0.0f;
}

__global__ void colstats_kernel(const float* __restrict__ y,
                                float* __restrict__ sums, float* __restrict__ sq,
                                int C, int rowsPerBlock)
{
    int c = threadIdx.x;                       // blockDim.x == C
    int r0 = blockIdx.x * rowsPerBlock;
    float s = 0.0f, q = 0.0f;
    for (int i = 0; i < rowsPerBlock; ++i) {
        float v = y[(r0 + i) * C + c];
        s += v; q += v * v;
    }
    atomicAdd(&sums[c], s);
    atomicAdd(&sq[c], q);
}

template <typename T>
__global__ void bn_apply(const float* __restrict__ x, const float* __restrict__ sums,
                         const float* __restrict__ sq, const float* __restrict__ gamma,
                         const float* __restrict__ beta, T* __restrict__ y,
                         int Cmask, float invN, int total)
{
    int i = blockIdx.x * blockDim.x + threadIdx.x;
    if (i >= total) return;
    int c = i & Cmask;
    float mean = sums[c] * invN;
    float var  = sq[c] * invN - mean * mean;
    float v = (x[i] - mean) * rsqrtf(var + BN_EPS) * gamma[c] + beta[c];
    y[i] = (T)v;
}

__global__ void cast_bf16_kernel(const float* __restrict__ x, bf16_t* __restrict__ y, int n4)
{
    int i = blockIdx.x * blockDim.x + threadIdx.x;
    if (i >= n4) return;
    v4f v = *(const v4f*)&x[i * 4];
    v4bf o;
#pragma unroll
    for (int j = 0; j < 4; ++j) o[j] = (bf16_t)v[j];
    *(v4bf*)&y[i * 4] = o;
}

// ---------------------------------------------------------------------------
// Host-side launch sequence
// ---------------------------------------------------------------------------
extern "C" void kernel_launch(void* const* d_in, const int* in_sizes, int n_in,
                              void* d_out, int out_size, void* d_ws, size_t ws_size,
                              hipStream_t stream)
{
    (void)in_sizes; (void)n_in; (void)out_size; (void)ws_size;

    const float* features = (const float*)d_in[0];
    const int*   adj      = (const int*)d_in[1];
    const float* kernel1  = (const float*)d_in[2];
    const float* bias1    = (const float*)d_in[3];
    const float* gamma1   = (const float*)d_in[4];
    const float* beta1    = (const float*)d_in[5];
    const float* kernel2  = (const float*)d_in[6];
    const float* bias2    = (const float*)d_in[7];
    const float* gamma2   = (const float*)d_in[8];
    const float* beta2    = (const float*)d_in[9];
    float* out = (float*)d_out;

    char* ws = (char*)d_ws;
    size_t off = 0;
    auto alloc = [&](size_t bytes) -> void* {
        void* p = ws + off;
        off += (bytes + 255) & ~(size_t)255;
        return p;
    };

    float*  mrow    = (float*)alloc((size_t)NROW * 4);
    float*  sinv    = (float*)alloc((size_t)NROW * 4);
    float*  pm      = (float*)alloc((size_t)NCHUNK * NROW * 4);
    float*  ps      = (float*)alloc((size_t)NCHUNK * NROW * 4);
    float*  stats   = (float*)alloc(768 * 4);      // [sums1|sq1|sums2|sq2]
    float*  sums1   = stats;
    float*  sq1     = stats + 256;
    float*  sums2   = stats + 512;
    float*  sq2     = stats + 640;
    float*  region0 = (float*)alloc((size_t)NROW * FDIM  * 4);
    float*  region1 = (float*)alloc((size_t)NROW * C0DIM * 4);
    float*  people0 = (float*)alloc((size_t)NROW * C0DIM * 4);
    float*  people1 = (float*)alloc((size_t)NROW * C1DIM * 4);
    bf16_t* Xb      = (bf16_t*)alloc((size_t)NROW * FDIM * 2);
    bf16_t* k1b     = (bf16_t*)alloc((size_t)FDIM * C0DIM * 2);
    bf16_t* k2b     = (bf16_t*)alloc((size_t)C0DIM * C1DIM * 2);
    bf16_t* r1b     = (bf16_t*)alloc((size_t)NROW * C0DIM * 2);

    // 0) zero BN partial-stat accumulators (fresh every call: graph-safe)
    zero_f32<<<3, 256, 0, stream>>>(stats, 768);

    // 1) masked-softmax row stats: chunked partials + deterministic merge
    softmax_stats_phase1<<<(NROW / 4) * NCHUNK / 256, 256, 0, stream>>>(adj, pm, ps);
    softmax_stats_phase2<<<NROW / 256, 256, 0, stream>>>(pm, ps, mrow, sinv);

    // 2) bf16 operand prep (X, K1, K2 live in L2 afterwards)
    cast_bf16_kernel<<<(NROW * FDIM / 4) / 256, 256, 0, stream>>>(features, Xb, NROW * FDIM / 4);
    cast_bf16_kernel<<<(FDIM * C0DIM / 4) / 256, 256, 0, stream>>>(kernel1, k1b, FDIM * C0DIM / 4);
    cast_bf16_kernel<<<(C0DIM * C1DIM / 4) / 256, 256, 0, stream>>>(kernel2, k2b, C0DIM * C1DIM / 4);

    // 3) region0 = w @ X          (fused softmax, WMMA, streams adj once)
    spmm_softmax_wmma<FDIM / 16, false><<<NROW / 64, 128, 0, stream>>>(
        adj, mrow, sinv, Xb, region0);

    // 4) region1 = region0 @ K1 + b1
    gemm_bias_wmma<C0DIM / 16><<<NROW / 64, 128, 0, stream>>>(
        region0, k1b, bias1, region1, FDIM);

    // 5) BN1 stats + apply (bf16 output feeds next WMMA)
    colstats_kernel<<<128, C0DIM, 0, stream>>>(region1, sums1, sq1, C0DIM, NROW / 128);
    bn_apply<bf16_t><<<(NROW * C0DIM) / 256, 256, 0, stream>>>(
        region1, sums1, sq1, gamma1, beta1, r1b, C0DIM - 1, 1.0f / NROW, NROW * C0DIM);

    // 6) people0 = w^T @ BN(region1)   (fused softmax, WMMA, streams adj once)
    spmm_softmax_wmma<C0DIM / 16, true><<<NROW / 64, 128, 0, stream>>>(
        adj, mrow, sinv, r1b, people0);

    // 7) people1 = people0 @ K2 + b2
    gemm_bias_wmma<C1DIM / 16><<<NROW / 64, 128, 0, stream>>>(
        people0, k2b, bias2, people1, C0DIM);

    // 8) BN2 stats + apply -> final f32 output
    colstats_kernel<<<128, C1DIM, 0, stream>>>(people1, sums2, sq2, C1DIM, NROW / 128);
    bn_apply<float><<<(NROW * C1DIM) / 256, 256, 0, stream>>>(
        people1, sums2, sq2, gamma2, beta2, out, C1DIM - 1, 1.0f / NROW, NROW * C1DIM);
}
